// ADFM_23218593202441
// MI455X (gfx1250) — matrix operand order repaired and checked by reading, not compiled, not run on hardware
//
#include <hip/hip_runtime.h>

// MI455X / gfx1250: wave32, WMMA 16x16x32 f16 (f32 accumulate), TDM staging.
// Pipeline: cvt(f32->f16) -> conv1x1 (TDM-staged B + WMMA)
//        -> conv3x3 offset (LDS-staged implicit-GEMM WMMA)
//        -> deformable conv (LDS-staged sampled B tile + WMMA)
//        -> conv1x1 (TDM-staged B + WMMA, f32 out)

typedef _Float16 half_t;
typedef __attribute__((ext_vector_type(16))) _Float16 v16h;
typedef __attribute__((ext_vector_type(8)))  _Float16 v8h;
typedef __attribute__((ext_vector_type(8)))  float    v8f;
typedef unsigned int u32x4 __attribute__((ext_vector_type(4)));
typedef int          i32x4 __attribute__((ext_vector_type(4)));
typedef int          i32x8 __attribute__((ext_vector_type(8)));

#define BB   4
#define CC   64
#define HH   192
#define WWID 192
#define HWP  (HH * WWID)          // 36864; divisible by 64 so 64-pixel tiles
                                  // never cross a batch image.
#define NPIX (BB * HWP)           // 147456
#define NT16 (NPIX / 16)          // 9216  16-pixel tiles
#define NT64 (NPIX / 64)          // 2304  64-pixel tiles
#define KDEF 576                  // 64 ch * 9 taps

// ---------------------------------------------------------------------------
// f32 -> f16 conversion
// ---------------------------------------------------------------------------
__global__ void cvt_f32_f16(const float* __restrict__ src, half_t* __restrict__ dst, int n) {
    int i = blockIdx.x * 256 + threadIdx.x;
    if (i < n) dst[i] = (half_t)src[i];
}

// ---------------------------------------------------------------------------
// Tensor Data Mover: 2D tile (tile_w elems x tile_h rows, row stride
// row_stride_elems, 2-byte elements) global -> contiguous LDS.
// D# per ISA 8.3/8.4: group0 {count=1, lds_addr, global_addr, type=2},
// group1 {data_size=1(2B), tensor_dim0/1, tile_dim0/1, tensor_dim0_stride}.
// ---------------------------------------------------------------------------
__device__ __forceinline__ void tdm_load_2d_f16(half_t* lds_dst, const half_t* gsrc,
                                                unsigned tile_w, unsigned tile_h,
                                                unsigned row_stride_elems) {
    unsigned long long ga = (unsigned long long)(uintptr_t)gsrc;
    unsigned lds_addr = (unsigned)(uintptr_t)lds_dst;   // flat->LDS: addr[31:0]
    u32x4 g0;
    g0[0] = 1u;                                         // count=1, user mode
    g0[1] = lds_addr;
    g0[2] = (unsigned)ga;
    g0[3] = (unsigned)(ga >> 32) | (2u << 30);          // addr[56:32] | type=2
    i32x8 g1;
    g1[0] = (int)(1u << 16);                            // data_size=1 (2B), mask=0
    g1[1] = (int)((row_stride_elems & 0xFFFFu) << 16);  // tensor_dim0[15:0]
    g1[2] = (int)(((tile_h & 0xFFFFu) << 16) |          // tensor_dim1[15:0]
                  ((row_stride_elems >> 16) & 0xFFFFu));// tensor_dim0[31:16]
    g1[3] = (int)((tile_w & 0xFFFFu) << 16);            // tile_dim0
    g1[4] = (int)(tile_h & 0xFFFFu);                    // tile_dim1 (tile_dim2=0)
    g1[5] = (int)row_stride_elems;                      // tensor_dim0_stride[31:0]
    g1[6] = 0;
    g1[7] = 0;
    i32x4 z4 = {0, 0, 0, 0};
#if __clang_major__ >= 23
    i32x8 z8 = {0, 0, 0, 0, 0, 0, 0, 0};
    __builtin_amdgcn_tensor_load_to_lds(g0, g1, z4, z4, z8, 0);
#else
    __builtin_amdgcn_tensor_load_to_lds(g0, g1, z4, z4, 0);
#endif
}

// ---------------------------------------------------------------------------
// WMMA A fragment: lane hl row m0+hl; halves s=0..15 -> K=hi*16+s
// (contiguous order, matches B): two adjacent 16B runs.
// ---------------------------------------------------------------------------
__device__ __forceinline__ v16h load_a_frag(const half_t* __restrict__ A, int ld,
                                            int m0, int k0, int lane) {
    int hl = lane & 15, hi = lane >> 4;
    const half_t* r = A + (size_t)(m0 + hl) * ld + k0 + hi * 16;
    v8h lo = *(const v8h*)(r);
    v8h hv = *(const v8h*)(r + 8);
    v16h a;
#pragma unroll
    for (int s = 0; s < 8; ++s) { a[s] = lo[s]; a[s + 8] = hv[s]; }
    return a;
}

__device__ __forceinline__ v16h load_a_frag_guard(const half_t* __restrict__ A, int ld,
                                                  int m0, int k0, int lane, int mlim) {
    int hl = lane & 15, hi = lane >> 4;
    int m = m0 + hl;
    v16h a;
    if (m < mlim) {
        const half_t* r = A + (size_t)m * ld + k0 + hi * 16;
        v8h lo = *(const v8h*)(r);
        v8h hv = *(const v8h*)(r + 8);
#pragma unroll
        for (int s = 0; s < 8; ++s) { a[s] = lo[s]; a[s + 8] = hv[s]; }
    } else {
#pragma unroll
        for (int s = 0; s < 16; ++s) a[s] = (half_t)0.f;
    }
    return a;
}

// ---------------------------------------------------------------------------
// conv1x1 as GEMM, block tile M=64 x N=64, K=64.
// Wave 0 TDM-stages B [k=64][n=64] into LDS; block transposes to [n][k]
// (k-stride 80 halves: 160B, 32B-aligned v16h reads); 8 waves x 2 subtiles,
// A fragments loaded once and reused across both N subtiles.
// ---------------------------------------------------------------------------
template <typename OutT>
__global__ void conv1x1_tdm(const half_t* __restrict__ in, const half_t* __restrict__ wgt,
                            const float* __restrict__ bias, OutT* __restrict__ out) {
    __shared__ __align__(32) half_t bstage[64 * 64];   // [k][n] (TDM dest)
    __shared__ __align__(32) half_t bt[64 * 80];       // [n][k] padded
    int t  = threadIdx.x;                              // 0..255
    int tn = blockIdx.x;                               // 64-pixel tile
    int b  = (tn * 64) / HWP, p0 = (tn * 64) % HWP;

    if (t < 32) {                                      // wave 0 drives the DMA
        tdm_load_2d_f16(bstage, in + (size_t)b * CC * HWP + p0,
                        64u, 64u, (unsigned)HWP);
        __builtin_amdgcn_s_wait_tensorcnt(0);
    }
    __syncthreads();

#pragma unroll
    for (int f = t; f < 64 * 64; f += 256) {           // transpose once
        int k = f >> 6, n = f & 63;
        bt[n * 80 + k] = bstage[f];
    }
    __syncthreads();

    int lane = t & 31, wv = t >> 5;
    int hl = lane & 15, hi = lane >> 4;
    int m0 = (wv & 3) * 16;
    v16h a0 = load_a_frag(wgt, CC, m0, 0,  lane);      // reused across subtiles
    v16h a1 = load_a_frag(wgt, CC, m0, 32, lane);
#pragma unroll
    for (int j = 0; j < 2; ++j) {
        int n0 = ((wv >> 2) + 2 * j) * 16;
        v8f acc;
#pragma unroll
        for (int i = 0; i < 8; ++i) acc[i] = bias[m0 + i + hi * 8];
        v16h b0 = *(const v16h*)&bt[(n0 + hl) * 80 + hi * 16];
        acc = __builtin_amdgcn_wmma_f32_16x16x32_f16(false, a0, false, b0,
                                                     (short)0, acc, false, false);
        v16h b1 = *(const v16h*)&bt[(n0 + hl) * 80 + 32 + hi * 16];
        acc = __builtin_amdgcn_wmma_f32_16x16x32_f16(false, a1, false, b1,
                                                     (short)0, acc, false, false);
        OutT* o = out + (size_t)b * CC * HWP + p0 + n0 + hl;
#pragma unroll
        for (int i = 0; i < 8; ++i)
            o[(size_t)(m0 + i + hi * 8) * HWP] = (OutT)acc[i];
    }
}

// ---------------------------------------------------------------------------
// 3x3 offset conv, implicit GEMM: M=18 (guarded), K=576, N=16 pixels/block.
// Block (128 thr) builds B tile [N][K] in LDS once (coalesced, zero-padded
// halo), then waves 0-1 run the 18-step WMMA loop with v16h LDS fragments.
// ---------------------------------------------------------------------------
__global__ void conv3x3_offset(const half_t* __restrict__ h, const half_t* __restrict__ w,
                               const float* __restrict__ bias, float* __restrict__ off) {
    __shared__ __align__(32) half_t samp[16 * KDEF];   // [N][k]
    int t  = threadIdx.x;                              // 0..127
    int tn = blockIdx.x;
    int b = (tn * 16) / HWP, p0 = (tn * 16) % HWP;
    int y = p0 / WWID, x0 = p0 - y * WWID;

    for (int f = t; f < KDEF * 16; f += 128) {         // N fastest => coalesced
        int k = f >> 4, N = f & 15;
        int c = k / 9, r = k - 9 * c;
        int yy = y + r / 3 - 1;
        int xx = x0 + N + (r - (r / 3) * 3) - 1;
        bool ok = (yy >= 0) & (yy < HH) & (xx >= 0) & (xx < WWID);
        samp[N * KDEF + k] = ok ? h[((size_t)b * CC + c) * HWP + yy * WWID + xx]
                                : (half_t)0.f;
    }
    __syncthreads();

    int lane = t & 31, wv = t >> 5;
    if (wv < 2) {                                      // 2 M-tiles cover M=18
        int hl = lane & 15, hi = lane >> 4;
        int m0 = wv * 16;
        v8f acc;
#pragma unroll
        for (int i = 0; i < 8; ++i) {
            int m = m0 + i + hi * 8;
            acc[i] = (m < 18) ? bias[m] : 0.f;
        }
#pragma unroll 1
        for (int kt = 0; kt < 18; ++kt) {
            v16h a  = load_a_frag_guard(w, KDEF, m0, kt * 32, lane, 18);
            v16h bf = *(const v16h*)&samp[hl * KDEF + kt * 32 + hi * 16];
            acc = __builtin_amdgcn_wmma_f32_16x16x32_f16(false, a, false, bf,
                                                         (short)0, acc, false, false);
        }
        float* ob = off + (size_t)b * 18 * HWP + p0 + hl;
#pragma unroll
        for (int i = 0; i < 8; ++i) {
            int m = m0 + i + hi * 8;
            if (m < 18) ob[(size_t)m * HWP] = acc[i];
        }
    }
}

// ---------------------------------------------------------------------------
// Deformable conv: one 16-pixel tile per 128-thread block (4 waves).
// Phase 0: bilinear taps (channel-invariant) -> LDS.
// Phase 1: sampled B tile [N][K=576] f16 -> LDS (v16h-aligned fragments).
// Phase 2: 4 waves x 18-step WMMA K-loop (16 out channels each).
// ---------------------------------------------------------------------------
__global__ void deform_gemm(const half_t* __restrict__ h, const float* __restrict__ off,
                            const half_t* __restrict__ w, const float* __restrict__ bias,
                            half_t* __restrict__ d) {
    __shared__ __align__(32) half_t samp[16 * KDEF];   // [N][k]
    __shared__ float swgt[9 * 16 * 4];
    __shared__ int   sidx[9 * 16 * 4];

    int t  = threadIdx.x;                              // 0..127
    int tn = blockIdx.x;
    int b = (tn * 16) / HWP, p0 = (tn * 16) % HWP;
    int y = p0 / WWID, x0 = p0 - y * WWID;

    __builtin_prefetch(w + (size_t)t * 64, 0, 1);      // warm w_def (global_prefetch_b8)

    if (t < 144) {                                     // 9 taps x 16 pixels
        int r = t >> 4, N = t & 15;
        int p = p0 + N;
        float dy = off[((size_t)b * 18 + 2 * r)     * HWP + p];
        float dx = off[((size_t)b * 18 + 2 * r + 1) * HWP + p];
        float py = (float)(y + (r / 3) - 1) + dy;
        float px = (float)(x0 + N + (r % 3) - 1) + dx;
        float y0f = floorf(py), x0f = floorf(px);
        float fy = py - y0f, fx = px - x0f;
        int iy0 = (int)y0f, ix0 = (int)x0f;
        int iy1 = iy0 + 1,  ix1 = ix0 + 1;
        int cy0 = min(max(iy0, 0), HH - 1),   cy1 = min(max(iy1, 0), HH - 1);
        int cx0 = min(max(ix0, 0), WWID - 1), cx1 = min(max(ix1, 0), WWID - 1);
        bool ry0 = (iy0 >= 0) & (iy0 < HH),   ry1 = (iy1 >= 0) & (iy1 < HH);
        bool rx0 = (ix0 >= 0) & (ix0 < WWID), rx1 = (ix1 >= 0) & (ix1 < WWID);
        int e = (r * 16 + N) * 4;
        sidx[e + 0] = cy0 * WWID + cx0;  swgt[e + 0] = (ry0 & rx0) ? (1.f - fy) * (1.f - fx) : 0.f;
        sidx[e + 1] = cy0 * WWID + cx1;  swgt[e + 1] = (ry0 & rx1) ? (1.f - fy) * fx         : 0.f;
        sidx[e + 2] = cy1 * WWID + cx0;  swgt[e + 2] = (ry1 & rx0) ? fy * (1.f - fx)         : 0.f;
        sidx[e + 3] = cy1 * WWID + cx1;  swgt[e + 3] = (ry1 & rx1) ? fy * fx                 : 0.f;
    }
    __syncthreads();

    for (int f = t; f < KDEF * 16; f += 128) {
        int k = f >> 4, N = f & 15;
        int c = k / 9, r = k - 9 * c;
        int e = (r * 16 + N) * 4;
        const half_t* hb = h + ((size_t)b * CC + c) * HWP;
        float v = swgt[e + 0] * (float)hb[sidx[e + 0]]
                + swgt[e + 1] * (float)hb[sidx[e + 1]]
                + swgt[e + 2] * (float)hb[sidx[e + 2]]
                + swgt[e + 3] * (float)hb[sidx[e + 3]];
        samp[N * KDEF + k] = (half_t)v;
    }
    __syncthreads();

    int lane = t & 31, wv = t >> 5;
    int hl = lane & 15, hi = lane >> 4;
    int m0 = wv * 16;
    v8f acc;
#pragma unroll
    for (int i = 0; i < 8; ++i) acc[i] = bias[m0 + i + hi * 8];
#pragma unroll 1
    for (int kt = 0; kt < 18; ++kt) {
        v16h a  = load_a_frag(w, KDEF, m0, kt * 32, lane);
        v16h bf = *(const v16h*)&samp[hl * KDEF + kt * 32 + hi * 16];
        acc = __builtin_amdgcn_wmma_f32_16x16x32_f16(false, a, false, bf,
                                                     (short)0, acc, false, false);
    }
    half_t* ob = d + (size_t)b * CC * HWP + p0 + hl;
#pragma unroll
    for (int i = 0; i < 8; ++i)
        ob[(size_t)(m0 + i + hi * 8) * HWP] = (half_t)acc[i];
}

// ---------------------------------------------------------------------------
extern "C" void kernel_launch(void* const* d_in, const int* in_sizes, int n_in,
                              void* d_out, int out_size, void* d_ws, size_t ws_size,
                              hipStream_t stream) {
    const float* x     = (const float*)d_in[0];
    const float* w1    = (const float*)d_in[1];
    const float* b1    = (const float*)d_in[2];
    const float* w_off = (const float*)d_in[3];
    const float* b_off = (const float*)d_in[4];
    const float* w_def = (const float*)d_in[5];
    const float* b_def = (const float*)d_in[6];
    const float* w2    = (const float*)d_in[7];
    const float* b2    = (const float*)d_in[8];

    const size_t NX   = (size_t)BB * CC * HWP;   // 9,437,184
    const size_t NOFF = (size_t)BB * 18 * HWP;   // 2,654,208

    char* ws = (char*)d_ws;
    size_t cur = 0;
    auto alloc = [&](size_t bytes) { size_t r = cur; cur += (bytes + 255) & ~(size_t)255; return r; };
    half_t* x16   = (half_t*)(ws + alloc(NX * 2));
    half_t* h16   = (half_t*)(ws + alloc(NX * 2));
    half_t* d16   = (half_t*)(ws + alloc(NX * 2));
    float*  offb  = (float*) (ws + alloc(NOFF * 4));
    half_t* w1h   = (half_t*)(ws + alloc(64 * 64 * 2));
    half_t* woffh = (half_t*)(ws + alloc(18 * KDEF * 2));
    half_t* wdefh = (half_t*)(ws + alloc(64 * KDEF * 2));
    half_t* w2h   = (half_t*)(ws + alloc(64 * 64 * 2));

    cvt_f32_f16<<<(int)((NX + 255) / 256), 256, 0, stream>>>(x, x16, (int)NX);
    cvt_f32_f16<<<16, 256, 0, stream>>>(w1, w1h, 64 * 64);
    cvt_f32_f16<<<41, 256, 0, stream>>>(w_off, woffh, 18 * KDEF);
    cvt_f32_f16<<<144, 256, 0, stream>>>(w_def, wdefh, 64 * KDEF);
    cvt_f32_f16<<<16, 256, 0, stream>>>(w2, w2h, 64 * 64);

    conv1x1_tdm<half_t><<<NT64, 256, 0, stream>>>(x16, w1h, b1, h16);
    conv3x3_offset<<<NT16, 128, 0, stream>>>(h16, woffh, b_off, offb);
    deform_gemm<<<NT16, 128, 0, stream>>>(h16, offb, wdefh, b_def, d16);
    conv1x1_tdm<float><<<NT64, 256, 0, stream>>>(d16, w2h, b2, (float*)d_out);
}
